// STAddAttention2_85899345920235
// MI455X (gfx1250) — matrix operand order repaired and checked by reading, not compiled, not run on hardware
//
#include <hip/hip_runtime.h>
#include <hip/hip_bf16.h>

// Problem constants (match reference)
#define Bc 4
#define Nc 32
#define Lc 256
#define Sc_ 256
#define Hc 8
#define Ec 64
#define Dc 64
#define LT 32          // L rows per block
#define THREADS 256    // 8 wave32s

typedef __attribute__((ext_vector_type(16))) __bf16 v16bf;
typedef __attribute__((ext_vector_type(2)))  __bf16 v2bf;
typedef __attribute__((ext_vector_type(8)))  float  v8f;

union FragBF {
    v16bf v;
    unsigned short u[16];
};

__device__ __forceinline__ unsigned short f2bf(float f) {
    unsigned u = __float_as_uint(f);
    unsigned r = u + 0x7FFFu + ((u >> 16) & 1u);   // round-to-nearest-even
    return (unsigned short)(r >> 16);
}

#if __has_builtin(__builtin_amdgcn_cvt_pk_bf16_f32)
__device__ __forceinline__ uint2 pack4bf(float4 f) {
    union { v2bf v; unsigned u; } a, b;
    a.v = __builtin_amdgcn_cvt_pk_bf16_f32(f.x, f.y);  // v_cvt_pk_bf16_f32
    b.v = __builtin_amdgcn_cvt_pk_bf16_f32(f.z, f.w);
    uint2 o; o.x = a.u; o.y = b.u;
    return o;
}
#else
__device__ __forceinline__ uint2 pack4bf(float4 f) {
    uint2 o;
    o.x = (unsigned)f2bf(f.x) | ((unsigned)f2bf(f.y) << 16);
    o.y = (unsigned)f2bf(f.z) | ((unsigned)f2bf(f.w) << 16);
    return o;
}
#endif

__global__ __launch_bounds__(THREADS)
void st_add_attention_kernel(const float* __restrict__ q,
                             const float* __restrict__ k,
                             const float* __restrict__ v,
                             const float* __restrict__ mask,
                             const float* __restrict__ prior,
                             float* __restrict__ scores_out,
                             float* __restrict__ v_out)
{
    // LDS staging (total 60 KB)
    __shared__ alignas(16) unsigned short Qs[LT * Ec];      // 4 KB  bf16 Q tile
    __shared__ alignas(16) unsigned short KVs[64 * 64];     // 8 KB  bf16 K or V chunk
    __shared__ alignas(16) float          Stile[LT * Sc_];  // 32 KB prior -> scores -> exp
    __shared__ alignas(16) unsigned short Pt[LT * Sc_];     // 16 KB bf16 probabilities

    const int bidx  = blockIdx.x;
    const int ltile = bidx & 7;          // L / LT = 8 tiles
    const int bnh   = bidx >> 3;         // 0 .. B*N*H-1
    const int h     = bnh & (Hc - 1);
    const int bn    = bnh >> 3;          // b*N + n
    const int l0    = ltile * LT;

    const int tid   = threadIdx.x;
    const int wave  = tid >> 5;
    const int lane  = tid & 31;
    const int nloc  = lane & 15;
    const int khalf = lane >> 4;

    // Base pointers (Q/K/V row stride = H*E = H*D = 512 floats, 16B-aligned rows)
    const float* qbase = q + ((size_t)(bn * Lc + l0) * Hc + h) * Ec;
    const float* kbase = k + ((size_t)bn * Sc_ * Hc + h) * Ec;
    const float* vbase = v + ((size_t)bn * Sc_ * Hc + h) * Dc;
    const float* mbase = mask + (size_t)l0 * Sc_;
    const float* pbase = prior + ((size_t)bnh * Lc + l0) * Sc_;  // contiguous 32 KB stripe
    float* sobase = scores_out + ((size_t)bnh * Lc + l0) * Sc_;
    float* obase  = v_out + ((size_t)(bn * Lc + l0) * Hc + h) * Dc;

    // ---- Async DMA: prior stripe (32 KB, contiguous) -> Stile in LDS --------
    // 2048 x 16B transfers, 8 per thread; tracked on ASYNCcnt.
    #pragma unroll
    for (int i = 0; i < 8; ++i) {
        int idx4 = tid + THREADS * i;
        unsigned ldsa = (unsigned)(size_t)&((float4*)Stile)[idx4];  // generic -> LDS offset
        unsigned goff = (unsigned)(idx4 * 16);
        asm volatile("global_load_async_to_lds_b128 %0, %1, %2"
                     :: "v"(ldsa), "v"(goff), "s"(pbase)
                     : "memory");
    }

    // ---- Load Q tile (32x64) as bf16: 512 float4 = 2 per thread -------------
    #pragma unroll
    for (int i = 0; i < 2; ++i) {
        int idx4 = tid + THREADS * i;                // uint2 index into Qs
        int r = idx4 >> 4, c4 = (idx4 & 15) * 4;
        float4 f = *(const float4*)&qbase[(size_t)r * (Hc * Ec) + c4];
        ((uint2*)Qs)[idx4] = pack4bf(f);
    }

    // ================= Phase 1: scores = Q K^T + mask + prior =================
    const int lsub = wave & 1;           // 16-row half of the 32-row tile
    const int ssub = wave >> 1;          // 16-col subtile within a 64-col chunk

    for (int sc = 0; sc < 4; ++sc) {     // stream K in 64-row S-chunks
        __syncthreads();                 // protect KVs reuse (and Qs on first iter)
        #pragma unroll
        for (int i = 0; i < 4; ++i) {    // 1024 float4 = 4 per thread
            int idx4 = tid + THREADS * i;
            int r = idx4 >> 4, c4 = (idx4 & 15) * 4;
            float4 f = *(const float4*)&kbase[(size_t)(sc * 64 + r) * (Hc * Ec) + c4];
            ((uint2*)KVs)[idx4] = pack4bf(f);
        }
        if (sc < 3) {                    // prefetch next K chunk (global_prefetch_b8)
            const char* pf = (const char*)&kbase[(size_t)((sc + 1) * 64 + (tid >> 2))
                                                 * (Hc * Ec) + (tid & 3) * 16];
            __builtin_prefetch(pf, 0, 1);
        }
        if (sc == 0)                     // prior DMA must land before first Stile read
            asm volatile("s_wait_asynccnt 0x0" ::: "memory");
        __syncthreads();

        v8f acc = {};
        #pragma unroll
        for (int kk = 0; kk < 2; ++kk) { // E = 64 -> two K=32 WMMAs
            FragBF fa, fb;
            const int mrow = lsub * 16 + nloc;
            const int srow = ssub * 16 + nloc;
            #pragma unroll
            for (int j = 0; j < 16; ++j) {
                // A (16x32 bf16): K = 16*khalf + j
                int ka = kk * 32 + khalf * 16 + j;
                fa.u[j] = Qs[mrow * Ec + ka];
                // B (32x16 bf16): column n = lane&15 (= s), K = 16*khalf + j (= e)
                int kb = kk * 32 + khalf * 16 + j;
                fb.u[j] = KVs[srow * 64 + kb];
            }
            acc = __builtin_amdgcn_wmma_f32_16x16x32_bf16(
                false, fa.v, false, fb.v, (short)0, acc, false, false);
        }

        // Epilogue: scores = acc + mask + prior (prior already resident in Stile)
        #pragma unroll
        for (int r = 0; r < 8; ++r) {
            int ml = lsub * 16 + khalf * 8 + r;      // C/D: row = r + 8*(lane>=16)
            int sl = sc * 64 + ssub * 16 + nloc;
            float val = acc[r] + mbase[(size_t)ml * Sc_ + sl]
                               + Stile[ml * Sc_ + sl];
            Stile[ml * Sc_ + sl] = val;
        }
    }

    // Prefetch first V chunk while softmax runs
    {
        const char* pf = (const char*)&vbase[(size_t)(tid >> 2) * (Hc * Dc)
                                             + (tid & 3) * 16];
        __builtin_prefetch(pf, 0, 1);
    }
    __syncthreads();

    // == Phase 2: write scores (b128), softmax over S, store bf16 P ===========
    {
        const float temp = 0.125f;                 // 1/sqrt(E)
        const int row = tid >> 3;                  // 32 rows, 8 lanes per row
        const int oct = tid & 7;
        float*  Srow = &Stile[row * Sc_ + oct * 32];      // exclusive 32-col segment
        float4* So4  = (float4*)(sobase + (size_t)row * Sc_ + oct * 32);
        uint2*  Prow = (uint2*)&Pt[row * Sc_ + oct * 32];

        float mx = -3.0e38f;
        #pragma unroll
        for (int j = 0; j < 8; ++j) {
            float4 x = ((const float4*)Srow)[j];
            So4[j] = x;                                // coalesced 128-bit scores store
            mx = fmaxf(mx, fmaxf(fmaxf(x.x, x.y), fmaxf(x.z, x.w)));
        }
        #pragma unroll
        for (int off = 1; off < 8; off <<= 1)
            mx = fmaxf(mx, __shfl_xor(mx, off, 32));   // row lanes share the wave

        float ssum = 0.0f;
        #pragma unroll
        for (int j = 0; j < 8; ++j) {
            float4 x = ((const float4*)Srow)[j];
            float4 e;
            e.x = __expf(temp * (x.x - mx));
            e.y = __expf(temp * (x.y - mx));
            e.z = __expf(temp * (x.z - mx));
            e.w = __expf(temp * (x.w - mx));
            ssum += (e.x + e.y) + (e.z + e.w);
            ((float4*)Srow)[j] = e;                    // stash exp in place
        }
        #pragma unroll
        for (int off = 1; off < 8; off <<= 1)
            ssum += __shfl_xor(ssum, off, 32);

        const float inv = 1.0f / ssum;
        #pragma unroll
        for (int j = 0; j < 8; ++j) {
            float4 e = ((const float4*)Srow)[j];
            e.x *= inv; e.y *= inv; e.z *= inv; e.w *= inv;
            Prow[j] = pack4bf(e);
        }
    }

    // ================= Phase 3: V_out = P (32x256) x V (256x64) ==============
    const int dsub = wave >> 1;                    // 2 L-subtiles x 4 D-subtiles
    v8f vacc = {};

    for (int sc = 0; sc < 4; ++sc) {               // stream V in 64-row S-chunks
        __syncthreads();                           // protect KVs (and Pt on first iter)
        #pragma unroll
        for (int i = 0; i < 4; ++i) {
            int idx4 = tid + THREADS * i;
            int r = idx4 >> 4, c4 = (idx4 & 15) * 4;
            float4 f = *(const float4*)&vbase[(size_t)(sc * 64 + r) * (Hc * Dc) + c4];
            ((uint2*)KVs)[idx4] = pack4bf(f);
        }
        if (sc < 3) {                              // prefetch next V chunk
            const char* pf = (const char*)&vbase[(size_t)((sc + 1) * 64 + (tid >> 2))
                                                 * (Hc * Dc) + (tid & 3) * 16];
            __builtin_prefetch(pf, 0, 1);
        }
        __syncthreads();

        #pragma unroll
        for (int kk = 0; kk < 2; ++kk) {
            FragBF fa, fb;
            const int mrow = lsub * 16 + nloc;
            const int dcol = dsub * 16 + nloc;
            #pragma unroll
            for (int j = 0; j < 16; ++j) {
                int ka = kk * 32 + khalf * 16 + j;
                fa.u[j] = Pt[mrow * Sc_ + sc * 64 + ka];   // A = probabilities
                int kb = kk * 32 + khalf * 16 + j;         // K index = s within chunk
                fb.u[j] = KVs[kb * 64 + dcol];             // B = V, column n = d
            }
            vacc = __builtin_amdgcn_wmma_f32_16x16x32_bf16(
                false, fa.v, false, fb.v, (short)0, vacc, false, false);
        }
    }

    // Store attention output [LT rows, stride H*D]
    #pragma unroll
    for (int r = 0; r < 8; ++r) {
        int ml = lsub * 16 + khalf * 8 + r;
        int d  = dsub * 16 + nloc;
        obase[(size_t)ml * (Hc * Dc) + d] = vacc[r];
    }
}

extern "C" void kernel_launch(void* const* d_in, const int* in_sizes, int n_in,
                              void* d_out, int out_size, void* d_ws, size_t ws_size,
                              hipStream_t stream) {
    (void)in_sizes; (void)n_in; (void)out_size; (void)d_ws; (void)ws_size;
    const float* q     = (const float*)d_in[0];
    const float* k     = (const float*)d_in[1];
    const float* v     = (const float*)d_in[2];
    const float* mask  = (const float*)d_in[3];
    const float* prior = (const float*)d_in[4];
    // d_in[5], d_in[6]: query/key lengths — full L/S in this config, unused.

    float* scores_out = (float*)d_out;
    float* v_out      = scores_out + (size_t)Bc * Nc * Hc * Lc * Sc_;

    dim3 grid(Bc * Nc * Hc * (Lc / LT));   // 8192 blocks
    dim3 block(THREADS);
    st_add_attention_kernel<<<grid, block, 0, stream>>>(q, k, v, mask, prior,
                                                        scores_out, v_out);
}
